// HipatchOur_58308476011173
// MI455X (gfx1250) — matrix operand order repaired and checked by reading, not compiled, not run on hardware
//
#include <hip/hip_runtime.h>
#include <hip/hip_bf16.h>

// Problem constants from the reference
#define Bn  16
#define Mn  24
#define Ln  32
#define Nn  64
#define Hn  64
#define LPn 32
#define SH  65    // padded LDS row stride for 64-wide rows
#define SZ  129   // padded LDS row stride for 128-wide rows

typedef __attribute__((ext_vector_type(16))) _Float16 v16h;
typedef __attribute__((ext_vector_type(8)))  float    v8f;

// ---- WMMA fragment helpers (wave32, v_wmma_f32_16x16x32_f16) -------------
// A (16x32, f16) layout: lanes 0-15 = rows M0..15 holding K = kb+0..15;
// lanes 16-31 = rows M0..15 holding K = kb+16..31 (matches B's contiguous order).
__device__ __forceinline__ v16h wmma_load_a_f32(const float* src, int stride,
                                                int row_base, int lane, int kbase) {
    const int r   = row_base + (lane & 15);
    const int off = kbase + ((lane >> 4) << 4);
    const float* p = src + r * stride + off;
    v16h a;
#pragma unroll
    for (int i = 0; i < 8; ++i) a[i]     = (_Float16)p[i];
#pragma unroll
    for (int i = 0; i < 8; ++i) a[8 + i] = (_Float16)p[8 + i];
    return a;
}

// B (32x16, f16) contiguous-k layout (this code's order): lane 0-15 = col N0..15 holding K = kb+0..15 packed;
// lanes 16-31 = same cols holding K = kb+16..31.  B[k][n] = W[n][k] (Linear: z @ W^T),
// so lane reads 16 contiguous halves of W row n starting at k — one aligned v16h.
__device__ __forceinline__ v16h wmma_load_b(const _Float16* __restrict__ W, int ldk,
                                            int col_base, int lane, int kbase) {
    const int c    = col_base + (lane & 15);
    const int koff = kbase + ((lane >> 4) << 4);
    return *(const v16h*)(W + c * ldk + koff);
}

__device__ __forceinline__ v8f wmma_mac(v16h a, v16h b, v8f c) {
    return __builtin_amdgcn_wmma_f32_16x16x32_f16(false, a, false, b,
                                                  (short)0, c, false, false);
}

// ---- Kernel 0: zero accumulators, convert weights to f16 ------------------
__global__ __launch_bounds__(256) void hp_prep_kernel(
    const float* __restrict__ gw1, const float* __restrict__ gw2,
    const float* __restrict__ dw1, const float* __restrict__ dw2,
    float* __restrict__ pooled, float* __restrict__ cnt,
    _Float16* __restrict__ w1h, _Float16* __restrict__ w2h,
    _Float16* __restrict__ dw1h, _Float16* __restrict__ dw2h)
{
    const int t = blockIdx.x * blockDim.x + threadIdx.x;
    if (t < Bn * Nn * Hn) pooled[t] = 0.0f;
    if (t < Bn * Nn)      cnt[t]    = 0.0f;
    if (t < Hn * Hn) {
        w1h[t]  = (_Float16)gw1[t];
        w2h[t]  = (_Float16)gw2[t];
        dw2h[t] = (_Float16)dw2[t];
    }
    if (t < Hn * 2 * Hn)  dw1h[t] = (_Float16)dw1[t];
}

// ---- One GCNII layer (inlined twice so weight pointers stay kernel-arg
//      derived -> global_load_b128, not flat_load) -------------------------
__device__ __forceinline__ void hp_gcn_layer(
    const float* __restrict__ gw, const float* __restrict__ gb,
    const _Float16* __restrict__ wh,
    float* __restrict__ h0buf, float* __restrict__ hcur, float* __restrict__ Gbuf,
    float* __restrict__ aggv, float* __restrict__ aggW,
    const float* __restrict__ mbuf, float deg,
    int tid, int lane, int m16)
{
    // masked mean over L (one channel per thread)
    {
        float s = 0.0f;
#pragma unroll
        for (int l = 0; l < Ln; ++l) s += mbuf[l] * hcur[l * SH + tid];
        aggv[tid] = s / deg;
    }
    __syncthreads();
    // aggW = agg @ W^T  (rank-1 part of GCN propagation), fp32
    {
        const float* wr = gw + tid * Hn;
        float s = 0.0f;
#pragma unroll
        for (int k = 0; k < Hn; ++k) s += aggv[k] * wr[k];
        aggW[tid] = s;
    }
    // G = h0 @ W^T via WMMA (A identical for both layers)
    {
        v8f acc[4];
#pragma unroll
        for (int nt = 0; nt < 4; ++nt) { v8f z = {0,0,0,0,0,0,0,0}; acc[nt] = z; }
#pragma unroll
        for (int kc = 0; kc < 2; ++kc) {
            const v16h a = wmma_load_a_f32(h0buf, SH, m16, lane, kc * 32);
#pragma unroll
            for (int nt = 0; nt < 4; ++nt) {
                const v16h bb = wmma_load_b(wh, Hn, nt * 16, lane, kc * 32);
                acc[nt] = wmma_mac(a, bb, acc[nt]);
            }
        }
        // scatter D: VGPR v, lanes 0-15 -> row m16+v, lanes 16-31 -> row m16+8+v
        const int colr = lane & 15;
        const int rb   = m16 + ((lane >> 4) << 3);
#pragma unroll
        for (int nt = 0; nt < 4; ++nt)
#pragma unroll
            for (int v = 0; v < 8; ++v)
                Gbuf[(rb + v) * SH + nt * 16 + colr] = acc[nt][v];
    }
    __syncthreads();
    // h = relu(0.9*mask*aggW + 0.1*G + bias) + h
    for (int e = tid; e < Ln * Hn; e += 64) {
        const int l = e >> 6, h = e & 63;
        const float pre = 0.9f * mbuf[l] * aggW[h] + 0.1f * Gbuf[l * SH + h] + gb[h];
        hcur[l * SH + h] += fmaxf(pre, 0.0f);
    }
    __syncthreads();
}

// ---- Kernel 1: fused per-patch encoder + 2 GCNII layers + masked pool -----
__global__ __launch_bounds__(64) void hp_gcn_patch_kernel(
    const float* __restrict__ X,    const float* __restrict__ TT,
    const float* __restrict__ MSK,
    const float* __restrict__ tesw, const float* __restrict__ tesb,
    const float* __restrict__ tepw, const float* __restrict__ tepb,
    const float* __restrict__ obsw, const float* __restrict__ obsb,
    const float* __restrict__ nodevec,
    const float* __restrict__ gw1,  const float* __restrict__ gb1,
    const float* __restrict__ gw2,  const float* __restrict__ gb2,
    const _Float16* __restrict__ w1h, const _Float16* __restrict__ w2h,
    float* __restrict__ pooled, float* __restrict__ cnt)
{
    __shared__ float h0buf[Ln * SH];   // h0 (A-matrix for both layer GEMMs)
    __shared__ float hcur[Ln * SH];    // running h
    __shared__ float Gbuf[Ln * SH];    // h0 @ W^T
    __shared__ float aggv[Hn];
    __shared__ float aggW[Hn];
    __shared__ float mbuf[Ln];
    __shared__ float xbuf[Ln];
    __shared__ float tbuf[Ln];
    __shared__ float degsh, cntsh;

    const int p   = blockIdx.x;            // b*N*M + n*M + m
    const int b   = p / (Nn * Mn);
    const int rem = p - b * Nn * Mn;
    const int n   = rem / Mn;
    const int m   = rem - n * Mn;
    const int tid  = threadIdx.x;
    const int lane = tid & 31;
    const int m16  = (tid >> 5) * 16;

    const float sw = tesw[0], sb = tesb[0];

    if (tid < Ln) {
        const int gidx = ((b * Mn + m) * Ln + tid) * Nn + n;
        mbuf[tid] = MSK[gidx];
        xbuf[tid] = X[gidx];
        tbuf[tid] = TT[gidx];
    }
    __syncthreads();
    if (tid == 0) {
        float c = 0.0f;
#pragma unroll
        for (int l = 0; l < Ln; ++l) c += mbuf[l];
        cntsh = c;
        degsh = fmaxf(c, 1.0f);
    }

    // h0 = relu(x*obs_w + obs_b + nodevec[n] + te(tt))
    for (int e = tid; e < Ln * Hn; e += 64) {
        const int l = e >> 6, h = e & 63;
        const float t   = tbuf[l];
        const float teh = (h == 0) ? (t * sw + sb)
                                   : __sinf(t * tepw[h - 1] + tepb[h - 1]);
        float v = xbuf[l] * obsw[h] + obsb[h] + nodevec[n * Hn + h] + teh;
        v = fmaxf(v, 0.0f);
        h0buf[l * SH + h] = v;
        hcur[l * SH + h]  = v;
    }
    __syncthreads();

    const float deg = degsh;
    hp_gcn_layer(gw1, gb1, w1h, h0buf, hcur, Gbuf, aggv, aggW, mbuf, deg, tid, lane, m16);
    hp_gcn_layer(gw2, gb2, w2h, h0buf, hcur, Gbuf, aggv, aggW, mbuf, deg, tid, lane, m16);

    // masked pool partial -> global accumulators
    {
        float s = 0.0f;
#pragma unroll
        for (int l = 0; l < Ln; ++l) s += mbuf[l] * hcur[l * SH + tid];
        atomicAdd(&pooled[(b * Nn + n) * Hn + tid], s);
        if (tid == 0) atomicAdd(&cnt[b * Nn + n], cntsh);
    }
}

// ---- Kernel 2: decoder MLP over (b,n) with WMMA ---------------------------
__global__ __launch_bounds__(64) void hp_decoder_kernel(
    const float* __restrict__ tsp,
    const float* __restrict__ tesw, const float* __restrict__ tesb,
    const float* __restrict__ tepw, const float* __restrict__ tepb,
    const float* __restrict__ pooled, const float* __restrict__ cnt,
    const _Float16* __restrict__ dw1h, const _Float16* __restrict__ dw2h,
    const float* __restrict__ db1, const float* __restrict__ db2,
    const float* __restrict__ dw3, const float* __restrict__ db3,
    float* __restrict__ out)
{
    __shared__ float zbuf[LPn * SZ];   // 32 x 128 (padded)
    __shared__ float ubuf[LPn * SH];
    __shared__ float vbuf[LPn * SH];
    __shared__ float pm[Hn];

    const int p = blockIdx.x;           // b*N + n
    const int b = p / Nn;
    const int n = p - b * Nn;
    const int tid  = threadIdx.x;
    const int lane = tid & 31;
    const int m16  = (tid >> 5) * 16;

    const float c = fmaxf(cnt[p], 1.0f);
    pm[tid] = pooled[p * Hn + tid] / c;
    const float sw = tesw[0], sb = tesb[0];
    __syncthreads();

    // z = [pooled_mean ; te(t_pred)]
    for (int e = tid; e < LPn * 128; e += 64) {
        const int l = e >> 7, ci = e & 127;
        float v;
        if (ci < Hn) {
            v = pm[ci];
        } else {
            const int h = ci - Hn;
            const float t = tsp[b * LPn + l];
            v = (h == 0) ? (t * sw + sb) : __sinf(t * tepw[h - 1] + tepb[h - 1]);
        }
        zbuf[l * SZ + ci] = v;
    }
    __syncthreads();

    const int colr = lane & 15;
    const int rb   = m16 + ((lane >> 4) << 3);

    // layer1: relu(z @ dw1^T + db1), K = 128
    {
        v8f acc[4];
#pragma unroll
        for (int nt = 0; nt < 4; ++nt) { v8f z = {0,0,0,0,0,0,0,0}; acc[nt] = z; }
#pragma unroll
        for (int kc = 0; kc < 4; ++kc) {
            const v16h a = wmma_load_a_f32(zbuf, SZ, m16, lane, kc * 32);
#pragma unroll
            for (int nt = 0; nt < 4; ++nt) {
                const v16h bb = wmma_load_b(dw1h, 2 * Hn, nt * 16, lane, kc * 32);
                acc[nt] = wmma_mac(a, bb, acc[nt]);
            }
        }
#pragma unroll
        for (int nt = 0; nt < 4; ++nt)
#pragma unroll
            for (int v = 0; v < 8; ++v) {
                const int col = nt * 16 + colr;
                ubuf[(rb + v) * SH + col] = fmaxf(acc[nt][v] + db1[col], 0.0f);
            }
    }
    __syncthreads();

    // layer2: relu(u @ dw2^T + db2), K = 64
    {
        v8f acc[4];
#pragma unroll
        for (int nt = 0; nt < 4; ++nt) { v8f z = {0,0,0,0,0,0,0,0}; acc[nt] = z; }
#pragma unroll
        for (int kc = 0; kc < 2; ++kc) {
            const v16h a = wmma_load_a_f32(ubuf, SH, m16, lane, kc * 32);
#pragma unroll
            for (int nt = 0; nt < 4; ++nt) {
                const v16h bb = wmma_load_b(dw2h, Hn, nt * 16, lane, kc * 32);
                acc[nt] = wmma_mac(a, bb, acc[nt]);
            }
        }
#pragma unroll
        for (int nt = 0; nt < 4; ++nt)
#pragma unroll
            for (int v = 0; v < 8; ++v) {
                const int col = nt * 16 + colr;
                vbuf[(rb + v) * SH + col] = fmaxf(acc[nt][v] + db2[col], 0.0f);
            }
    }
    __syncthreads();

    // head: scalar per (lp); out layout [1,B,LP,N]
    if (tid < LPn) {
        float s = db3[0];
#pragma unroll
        for (int h = 0; h < Hn; ++h) s += vbuf[tid * SH + h] * dw3[h];
        out[(b * LPn + tid) * Nn + n] = s;
    }
}

extern "C" void kernel_launch(void* const* d_in, const int* in_sizes, int n_in,
                              void* d_out, int out_size, void* d_ws, size_t ws_size,
                              hipStream_t stream) {
    (void)in_sizes; (void)n_in; (void)out_size; (void)ws_size;
    const float* tsp  = (const float*)d_in[0];
    const float* X    = (const float*)d_in[1];
    const float* TT   = (const float*)d_in[2];
    const float* MSK  = (const float*)d_in[3];
    const float* tesw = (const float*)d_in[4];
    const float* tesb = (const float*)d_in[5];
    const float* tepw = (const float*)d_in[6];
    const float* tepb = (const float*)d_in[7];
    const float* obsw = (const float*)d_in[8];
    const float* obsb = (const float*)d_in[9];
    const float* ndv  = (const float*)d_in[10];
    const float* gw1  = (const float*)d_in[11];
    const float* gb1  = (const float*)d_in[12];
    const float* gw2  = (const float*)d_in[13];
    const float* gb2  = (const float*)d_in[14];
    const float* dw1  = (const float*)d_in[15];
    const float* db1  = (const float*)d_in[16];
    const float* dw2  = (const float*)d_in[17];
    const float* db2  = (const float*)d_in[18];
    const float* dw3  = (const float*)d_in[19];
    const float* db3  = (const float*)d_in[20];
    float* out = (float*)d_out;

    // workspace layout (256B-aligned chunks)
    char* ws = (char*)d_ws;
    float*    pooled = (float*)(ws);               // 65536 f32
    float*    cntb   = (float*)(ws + 262144);      // 1024  f32
    _Float16* w1h    = (_Float16*)(ws + 266240);   // 4096  f16
    _Float16* w2h    = (_Float16*)(ws + 274432);   // 4096  f16
    _Float16* dw1h   = (_Float16*)(ws + 282624);   // 8192  f16
    _Float16* dw2h   = (_Float16*)(ws + 299008);   // 4096  f16

    hp_prep_kernel<<<256, 256, 0, stream>>>(gw1, gw2, dw1, dw2,
                                            pooled, cntb, w1h, w2h, dw1h, dw2h);

    hp_gcn_patch_kernel<<<Bn * Nn * Mn, 64, 0, stream>>>(
        X, TT, MSK, tesw, tesb, tepw, tepb, obsw, obsb, ndv,
        gw1, gb1, gw2, gb2, w1h, w2h, pooled, cntb);

    hp_decoder_kernel<<<Bn * Nn, 64, 0, stream>>>(
        tsp, tesw, tesb, tepw, tepb, pooled, cntb,
        dw1h, dw2h, db1, db2, dw3, db3, out);
}
